// TimeLSTM_9826885173874
// MI455X (gfx1250) — compile-verified, matrix operand order checked by reading, not run on hardware
//
#include <hip/hip_runtime.h>

// ---------------------------------------------------------------------------
// Time-LSTM for MI455X (gfx1250, wave32, WMMA bf16 path)
// B=64, S=512, I=H=768
// - input-projection GEMM: 16x64 tile, double-buffered loads, sized to fit the
//   128-VGPR budget even if occupancy hints are ignored (no scratch spills)
// - sequential scan: ONE persistent kernel (24 blocks), atomic grid barriers
//   between phases instead of 1024 kernel launches
// ---------------------------------------------------------------------------

typedef __bf16 bf16_t;
typedef __attribute__((ext_vector_type(16))) __bf16 v16bf;
typedef __attribute__((ext_vector_type(8)))  __bf16 v8bf;
typedef __attribute__((ext_vector_type(8)))  float  v8f;

#define TL_B 64
#define TL_S 512
#define TL_I 768
#define TL_H 768
#define TL_G4 (4 * TL_H)
#define TL_M  (TL_B * TL_S)
#define TL_SCAN_BLOCKS 24

// ---------------------------------------------------------------------------
// Fragment loader per CDNA5 ISA 7.12.2 (16-bit A/B layout, wave32):
//   lanes 0-15 : K = kBase+0..7   (VGPR0-3) and kBase+16..23 (VGPR4-7)
//   lanes 16-31: K = kBase+8..15  (VGPR0-3) and kBase+24..31 (VGPR4-7)
// ---------------------------------------------------------------------------
__device__ __forceinline__ v16bf tl_load_frag(const bf16_t* __restrict__ base,
                                              int ld, int rowBase, int kBase,
                                              int lane) {
  const int idx = lane & 15;
  const int hl  = lane >> 4;
  const bf16_t* p = base + (long)(rowBase + idx) * ld + kBase + hl * 8;
  v8bf g0 = *(const v8bf*)(p);        // K group +0
  v8bf g1 = *(const v8bf*)(p + 16);   // K group +16
  return __builtin_shufflevector(g0, g1, 0, 1, 2, 3, 4, 5, 6, 7,
                                         8, 9, 10, 11, 12, 13, 14, 15);
}

#define TL_WMMA(A, Bm, C) \
  __builtin_amdgcn_wmma_f32_16x16x32_bf16(false, (A), false, (Bm), (short)0, (C), false, false)

__device__ __forceinline__ float tl_sigmoid(float x) {
  return 1.0f / (1.0f + __expf(-x));
}

// ---------------------------------------------------------------------------
// Utility kernels
// ---------------------------------------------------------------------------
__global__ void tl_f32_to_bf16(const float* __restrict__ src,
                               bf16_t* __restrict__ dst, int n) {
  int i = blockIdx.x * blockDim.x + threadIdx.x;
  int stride = gridDim.x * blockDim.x;
  for (; i < n; i += stride) dst[i] = (bf16_t)src[i];
}

__global__ void tl_init_state(bf16_t* __restrict__ h0,
                              bf16_t* __restrict__ c_bf,
                              float* __restrict__ c_f32,
                              unsigned* __restrict__ sync, int n) {
  int i = blockIdx.x * blockDim.x + threadIdx.x;
  if (i < n) {
    h0[i]    = (bf16_t)0.0f;
    c_bf[i]  = (bf16_t)0.0f;
    c_f32[i] = 0.0f;
  }
  if (i == 0) *sync = 0u;   // reset grid-barrier counter every call (replay-safe)
}

// ---------------------------------------------------------------------------
// u_proj = inputs[BS,I] @ U_all_w[4H,I]^T + U_all_b  -> bf16 [BS,4H]
// One wave computes a 16x64 tile (1 A-frag x 4 B-frags = 4 WMMAs per k-step),
// software-pipelined. Register demand: 32 (acc) + 80 (2 frag sets) + addr
// ~= 124 VGPRs -> fits the default 128-VGPR cap with NO scratch spills.
// Wave jobs = (BS/16) * (4H/64) = 2048 * 48 = 98304 -> 12288 blocks x 256.
// ---------------------------------------------------------------------------
__global__ void __launch_bounds__(256, 2)
tl_uproj_gemm(const bf16_t* __restrict__ X,
              const bf16_t* __restrict__ W,
              const float* __restrict__ bias,
              bf16_t* __restrict__ Y) {
  const int K = TL_I;
  const int N = TL_G4;
  const int wave = (blockIdx.x * blockDim.x + threadIdx.x) >> 5;
  const int lane = threadIdx.x & 31;
  const int nQuads = N / 64;              // 48
  const int q  = wave % nQuads;
  const int mT = wave / nQuads;           // 0..2047
  const int rowBase = mT * 16;
  const int colBase = q * 64;

  v8f acc0 = {}, acc1 = {}, acc2 = {}, acc3 = {};

  v16bf ac  = tl_load_frag(X, K, rowBase, 0, lane);
  v16bf b0c = tl_load_frag(W, K, colBase + 0,  0, lane);
  v16bf b1c = tl_load_frag(W, K, colBase + 16, 0, lane);
  v16bf b2c = tl_load_frag(W, K, colBase + 32, 0, lane);
  v16bf b3c = tl_load_frag(W, K, colBase + 48, 0, lane);

  for (int k = 32; k < K; k += 32) {
    // prefetch next k-step while current WMMAs execute
    v16bf an  = tl_load_frag(X, K, rowBase, k, lane);
    v16bf b0n = tl_load_frag(W, K, colBase + 0,  k, lane);
    v16bf b1n = tl_load_frag(W, K, colBase + 16, k, lane);
    v16bf b2n = tl_load_frag(W, K, colBase + 32, k, lane);
    v16bf b3n = tl_load_frag(W, K, colBase + 48, k, lane);

    acc0 = TL_WMMA(ac, b0c, acc0);
    acc1 = TL_WMMA(ac, b1c, acc1);
    acc2 = TL_WMMA(ac, b2c, acc2);
    acc3 = TL_WMMA(ac, b3c, acc3);

    ac = an; b0c = b0n; b1c = b1n; b2c = b2n; b3c = b3n;
  }
  acc0 = TL_WMMA(ac, b0c, acc0);
  acc1 = TL_WMMA(ac, b1c, acc1);
  acc2 = TL_WMMA(ac, b2c, acc2);
  acc3 = TL_WMMA(ac, b3c, acc3);

  const int idx = lane & 15;
  const int hl  = lane >> 4;
  v8f accs[4] = {acc0, acc1, acc2, acc3};
#pragma unroll
  for (int s = 0; s < 4; ++s) {
    const int n = colBase + s * 16 + idx;
    const float bv = bias[n];
#pragma unroll
    for (int r = 0; r < 8; ++r) {
      const int row = rowBase + r + hl * 8;   // C/D layout: VGPR r -> M = r + 8*half
      Y[(long)row * N + n] = (bf16_t)(accs[s][r] + bv);
    }
  }
}

// ---------------------------------------------------------------------------
// Monotonic grid barrier for the persistent scan kernel (24 resident blocks).
// ---------------------------------------------------------------------------
__device__ __forceinline__ void tl_grid_barrier(unsigned* __restrict__ sync,
                                                unsigned target) {
  __syncthreads();
  if (threadIdx.x == 0) {
    __threadfence();   // release prior global writes
    __hip_atomic_fetch_add(sync, 1u, __ATOMIC_RELEASE, __HIP_MEMORY_SCOPE_AGENT);
    while (__hip_atomic_load(sync, __ATOMIC_ACQUIRE, __HIP_MEMORY_SCOPE_AGENT) < target) {
      __builtin_amdgcn_s_sleep(2);
    }
    __threadfence();   // acquire other blocks' writes
  }
  __syncthreads();
}

// ---------------------------------------------------------------------------
// Persistent scan kernel: all 512 steps in one launch.
// 24 blocks x 8 waves = 192 wave jobs = (B/16) x (H/16) tiles; each wave keeps
// the same (batch-tile, hidden-tile) for the whole sequence.
// Per step:
//   phase A: c_s1 = tanh(c @ W_d^T + b_d); c_adj = c - c_s1 + c_s1*ts
//   <grid barrier>
//   phase B: outs = h @ W_all^T + b + u_t; gates; update h,c; write out[:,t,:]
//   <grid barrier>
// ---------------------------------------------------------------------------
__global__ void __launch_bounds__(256, 1)
tl_scan_persistent(const bf16_t* __restrict__ Wd,
                   const float* __restrict__ Wd_b,
                   const bf16_t* __restrict__ Wall,
                   const float* __restrict__ Wall_b,
                   const bf16_t* __restrict__ uproj,
                   const float* __restrict__ timestamps,
                   bf16_t* __restrict__ h0,
                   bf16_t* __restrict__ h1,
                   bf16_t* __restrict__ c_bf,
                   float* __restrict__ c_f32,
                   float* __restrict__ c_adj,
                   unsigned* __restrict__ sync,
                   float* __restrict__ out) {
  const int K = TL_H;
  const int wave = (blockIdx.x << 3) + (threadIdx.x >> 5);  // 0..191
  const int lane = threadIdx.x & 31;
  const int nT = TL_H / 16;               // 48
  const int ct = wave % nT;
  const int rt = wave / nT;               // 0..3
  const int rowBase = rt * 16;
  const int colBase = ct * 16;
  const int idx = lane & 15;
  const int hl  = lane >> 4;
  const int j = colBase + idx;

  const float bdj = Wd_b[j];
  const float bF = Wall_b[0 * TL_H + j];
  const float bI = Wall_b[1 * TL_H + j];
  const float bO = Wall_b[2 * TL_H + j];
  const float bC = Wall_b[3 * TL_H + j];

  unsigned barrier_no = 0;

  for (int t = 0; t < TL_S; ++t) {
    const bf16_t* h_in = (t & 1) ? h1 : h0;
    bf16_t* h_out      = (t & 1) ? h0 : h1;

    // ---------------- phase A: c_adj ----------------
    {
      v8f acc = {};
      v16bf ac = tl_load_frag(c_bf, K, rowBase, 0, lane);
      v16bf bc = tl_load_frag(Wd,   K, colBase, 0, lane);
      for (int k = 32; k < K; k += 32) {
        v16bf an = tl_load_frag(c_bf, K, rowBase, k, lane);
        v16bf bn = tl_load_frag(Wd,   K, colBase, k, lane);
        acc = TL_WMMA(ac, bc, acc);
        ac = an; bc = bn;
      }
      acc = TL_WMMA(ac, bc, acc);

#pragma unroll
      for (int r = 0; r < 8; ++r) {
        const int bi = rowBase + r + hl * 8;
        const float s1 = tanhf(acc[r] + bdj);
        const float cv = c_f32[bi * TL_H + j];
        const float ts = timestamps[bi * TL_S + t];
        c_adj[bi * TL_H + j] = cv - s1 + s1 * ts;
      }
    }

    tl_grid_barrier(sync, (++barrier_no) * TL_SCAN_BLOCKS);

    // ---------------- phase B: gates + state update ----------------
    {
      v8f accF = {}, accI = {}, accO = {}, accC = {};
      v16bf ac  = tl_load_frag(h_in, K, rowBase, 0, lane);
      v16bf bFc = tl_load_frag(Wall, K, 0 * TL_H + colBase, 0, lane);
      v16bf bIc = tl_load_frag(Wall, K, 1 * TL_H + colBase, 0, lane);
      v16bf bOc = tl_load_frag(Wall, K, 2 * TL_H + colBase, 0, lane);
      v16bf bCc = tl_load_frag(Wall, K, 3 * TL_H + colBase, 0, lane);
      for (int k = 32; k < K; k += 32) {
        v16bf an  = tl_load_frag(h_in, K, rowBase, k, lane);
        v16bf bFn = tl_load_frag(Wall, K, 0 * TL_H + colBase, k, lane);
        v16bf bIn = tl_load_frag(Wall, K, 1 * TL_H + colBase, k, lane);
        v16bf bOn = tl_load_frag(Wall, K, 2 * TL_H + colBase, k, lane);
        v16bf bCn = tl_load_frag(Wall, K, 3 * TL_H + colBase, k, lane);
        accF = TL_WMMA(ac, bFc, accF);
        accI = TL_WMMA(ac, bIc, accI);
        accO = TL_WMMA(ac, bOc, accO);
        accC = TL_WMMA(ac, bCc, accC);
        ac = an; bFc = bFn; bIc = bIn; bOc = bOn; bCc = bCn;
      }
      accF = TL_WMMA(ac, bFc, accF);
      accI = TL_WMMA(ac, bIc, accI);
      accO = TL_WMMA(ac, bOc, accO);
      accC = TL_WMMA(ac, bCc, accC);

#pragma unroll
      for (int r = 0; r < 8; ++r) {
        const int bi = rowBase + r + hl * 8;
        const long ub = ((long)bi * TL_S + t) * TL_G4;
        const float fg = tl_sigmoid(accF[r] + bF + (float)uproj[ub + 0 * TL_H + j]);
        const float ig = tl_sigmoid(accI[r] + bI + (float)uproj[ub + 1 * TL_H + j]);
        const float og = tl_sigmoid(accO[r] + bO + (float)uproj[ub + 2 * TL_H + j]);
        const float cg = tl_sigmoid(accC[r] + bC + (float)uproj[ub + 3 * TL_H + j]);
        const float ca = c_adj[bi * TL_H + j];
        const float cn = fg * ca + ig * cg;
        const float hn = og * tanhf(cn);
        c_f32[bi * TL_H + j] = cn;
        c_bf [bi * TL_H + j] = (bf16_t)cn;
        h_out[bi * TL_H + j] = (bf16_t)hn;
        out[((long)bi * TL_S + t) * TL_H + j] = hn;
      }
    }

    tl_grid_barrier(sync, (++barrier_no) * TL_SCAN_BLOCKS);
  }
}

// ---------------------------------------------------------------------------
// Host-side launcher (graph-capture safe: only stream launches, ws scratch).
// ---------------------------------------------------------------------------
extern "C" void kernel_launch(void* const* d_in, const int* in_sizes, int n_in,
                              void* d_out, int out_size, void* d_ws, size_t ws_size,
                              hipStream_t stream) {
  const float* inputs     = (const float*)d_in[0];  // [B,S,I]
  const float* timestamps = (const float*)d_in[1];  // [B,S]
  const float* Wall_w     = (const float*)d_in[2];  // [4H,H]
  const float* Wall_b     = (const float*)d_in[3];  // [4H]
  const float* Uall_w     = (const float*)d_in[4];  // [4H,I]
  const float* Uall_b     = (const float*)d_in[5];  // [4H]
  const float* Wd_w       = (const float*)d_in[6];  // [H,H]
  const float* Wd_b       = (const float*)d_in[7];  // [H]
  float* out = (float*)d_out;                       // [B,S,H]

  // Workspace layout (all offsets 256B-aligned)
  char* ws = (char*)d_ws;
  size_t off = 0;
  bf16_t* X_bf    = (bf16_t*)(ws + off); off += (size_t)TL_M * TL_I * 2;   // 48 MB
  bf16_t* Uall_bf = (bf16_t*)(ws + off); off += (size_t)TL_G4 * TL_I * 2;  // 4.5 MB
  bf16_t* Wall_bf = (bf16_t*)(ws + off); off += (size_t)TL_G4 * TL_H * 2;  // 4.5 MB
  bf16_t* Wd_bf   = (bf16_t*)(ws + off); off += (size_t)TL_H * TL_H * 2;   // 1.1 MB
  bf16_t* uproj   = (bf16_t*)(ws + off); off += (size_t)TL_M * TL_G4 * 2;  // 192 MB
  bf16_t* h_bf0   = (bf16_t*)(ws + off); off += (size_t)TL_B * TL_H * 2;
  bf16_t* h_bf1   = (bf16_t*)(ws + off); off += (size_t)TL_B * TL_H * 2;
  bf16_t* c_bf    = (bf16_t*)(ws + off); off += (size_t)TL_B * TL_H * 2;
  float*  c_f32   = (float*)(ws + off);  off += (size_t)TL_B * TL_H * 4;
  float*  c_adj   = (float*)(ws + off);  off += (size_t)TL_B * TL_H * 4;
  unsigned* sync  = (unsigned*)(ws + off); off += 256;

  // 1) convert weights + inputs to bf16, zero recurrent state + barrier counter
  tl_f32_to_bf16<<<4096, 256, 0, stream>>>(inputs, X_bf, TL_M * TL_I);
  tl_f32_to_bf16<<<2048, 256, 0, stream>>>(Uall_w, Uall_bf, TL_G4 * TL_I);
  tl_f32_to_bf16<<<2048, 256, 0, stream>>>(Wall_w, Wall_bf, TL_G4 * TL_H);
  tl_f32_to_bf16<<<1024, 256, 0, stream>>>(Wd_w, Wd_bf, TL_H * TL_H);
  tl_init_state<<<(TL_B * TL_H + 255) / 256, 256, 0, stream>>>(h_bf0, c_bf, c_f32, sync,
                                                               TL_B * TL_H);

  // 2) big parallel GEMM: u_proj (98304 wave jobs, 8 waves/block)
  tl_uproj_gemm<<<12288, 256, 0, stream>>>(X_bf, Uall_bf, Uall_b, uproj);

  // 3) sequential scan: single persistent kernel, grid barriers between phases
  tl_scan_persistent<<<TL_SCAN_BLOCKS, 256, 0, stream>>>(
      Wd_bf, Wd_b, Wall_bf, Wall_b, uproj, timestamps,
      h_bf0, h_bf1, c_bf, c_f32, c_adj, sync, out);
}
